// HSTUBlock_81664508166289
// MI455X (gfx1250) — matrix operand order repaired and checked by reading, not compiled, not run on hardware
//
#include <hip/hip_runtime.h>

// ---------------------------------------------------------------------------
// HSTU block for MI455X (gfx1250): bf16 WMMA GEMMs, f32 accumulate.
// B=4, L=2048, D=1024, N=8 heads, H=128.
// ---------------------------------------------------------------------------

#define B_  4
#define L_  2048
#define D_  1024
#define NH_ 8
#define HD_ 128
#define M_  (B_ * L_)          // 8192 tokens
#define EPS_ 1e-6f

typedef __attribute__((ext_vector_type(16))) __bf16 v16bf;
typedef __attribute__((ext_vector_type(8)))  float  v8f;

union FragU { uint4 u[2]; v16bf v; };

// ---- fragment loaders (16-byte aligned uint4 loads) -----------------------
// A-matrix 16x32 bf16 (MxK): lane l -> row l&15;
//   VGPR0..3: K = (l>=16?8:0)+{0..7}; VGPR4..7: K = 16+(l>=16?8:0)+{0..7}
__device__ __forceinline__ v16bf load_frag_a(const __bf16* base, int ld, int lane) {
  const int row = lane & 15;
  const int kb  = (lane >> 4) * 8;
  FragU f;
  f.u[0] = *reinterpret_cast<const uint4*>(base + (size_t)row * ld + kb);
  f.u[1] = *reinterpret_cast<const uint4*>(base + (size_t)row * ld + kb + 16);
  return f.v;
}

// B-matrix 32x16 bf16 (KxN), source stored transposed as [N][K] row-major:
// lane l -> col l&15; VGPR0..3: K = (l>=16?8:0)+{0..7}; VGPR4..7: K = 16+(l>=16?8:0)+{0..7}
__device__ __forceinline__ v16bf load_frag_b(const __bf16* baseT, int ld, int lane) {
  const int col = lane & 15;
  const int kb  = (lane >> 4) * 8;
  const uint4* p = reinterpret_cast<const uint4*>(baseT + (size_t)col * ld + kb);
  FragU f;
  f.u[0] = p[0];
  f.u[1] = p[2];
  return f.v;
}

__device__ __forceinline__ v8f wmma_bf16(v16bf a, v16bf b, v8f c) {
  return __builtin_amdgcn_wmma_f32_16x16x32_bf16(
      /*neg_a=*/false, a, /*neg_b=*/false, b,
      /*c_mod=*/(short)0, c, /*reuse_a=*/false, /*reuse_b=*/false);
}

__device__ __forceinline__ v8f vzero8() {
  v8f z;
#pragma unroll
  for (int i = 0; i < 8; ++i) z[i] = 0.0f;
  return z;
}

__device__ __forceinline__ float silu_f(float x) {
  return x / (1.0f + __expf(-x));
}

__device__ __forceinline__ float wave_sum(float v) {
#pragma unroll
  for (int off = 16; off > 0; off >>= 1) v += __shfl_xor(v, off, 32);
  return v;
}

// wave-local LDS RAW fence (wave-private LDS slabs; no workgroup barrier)
__device__ __forceinline__ void lds_sync_wave() {
  __builtin_amdgcn_wave_barrier();
#if __has_builtin(__builtin_amdgcn_s_wait_dscnt)
  __builtin_amdgcn_s_wait_dscnt(0);
#else
  asm volatile("s_wait_dscnt 0" ::: "memory");
#endif
  __builtin_amdgcn_wave_barrier();
}

// ---------------------------------------------------------------------------
// K0: transpose + convert weights to bf16.  mats 0..3 = wu,wq,wk,wv ([D][D]
// row-major, cols = n*H+h) -> wt4[mat][col][d].  mat 4 = wo ([nh][d]) ->
// wo_t[d][nh].  Generic out[c][d] = in[d][c].
// ---------------------------------------------------------------------------
__global__ __launch_bounds__(256) void k_wconv(
    const float* __restrict__ wu, const float* __restrict__ wq,
    const float* __restrict__ wk, const float* __restrict__ wv,
    const float* __restrict__ wo,
    __bf16* __restrict__ wt4, __bf16* __restrict__ wo_t) {
  const int idx = blockIdx.x * 256 + threadIdx.x;   // 0 .. 1M-1
  const int mat = blockIdx.y;                       // 0..4
  const int c = idx >> 10;
  const int d = idx & 1023;
  const float* src = (mat == 0) ? wu : (mat == 1) ? wq : (mat == 2) ? wk
                   : (mat == 3) ? wv : wo;
  __bf16* dst = (mat < 4) ? (wt4 + (size_t)mat * D_ * D_) : wo_t;
  dst[(size_t)c * D_ + d] = (__bf16)src[(size_t)d * D_ + c];
}

// ---------------------------------------------------------------------------
// K1: input LayerNorm over D -> bf16 x.  One block (256 thr) per token.
// ---------------------------------------------------------------------------
__global__ __launch_bounds__(256) void k_ln_in(
    const float* __restrict__ x, const float* __restrict__ gamma,
    const float* __restrict__ beta, __bf16* __restrict__ out) {
  const int m = blockIdx.x;
  const int tid = threadIdx.x, lane = tid & 31, w = tid >> 5;
  const float4 v = reinterpret_cast<const float4*>(x + (size_t)m * D_)[tid];
  float s  = v.x + v.y + v.z + v.w;
  float sq = v.x * v.x + v.y * v.y + v.z * v.z + v.w * v.w;
  s = wave_sum(s); sq = wave_sum(sq);
  __shared__ float rs[8], rq[8];
  if (lane == 0) { rs[w] = s; rq[w] = sq; }
  __syncthreads();
  float tot = 0.f, totq = 0.f;
#pragma unroll
  for (int i = 0; i < 8; ++i) { tot += rs[i]; totq += rq[i]; }
  const float mean = tot * (1.0f / D_);
  const float rstd = rsqrtf(totq * (1.0f / D_) - mean * mean + EPS_);
  const float vals[4] = {v.x, v.y, v.z, v.w};
#pragma unroll
  for (int e = 0; e < 4; ++e) {
    const int i = tid * 4 + e;
    out[(size_t)m * D_ + i] = (__bf16)((vals[e] - mean) * rstd * gamma[i] + beta[i]);
  }
}

// ---------------------------------------------------------------------------
// K2: fused u/q/k/v projections: silu(x @ W).  Block = 128x128 out tile,
// 8 waves, each wave 32x64 (2x4 WMMA accumulators).  wsel 3 (v) is stored
// transposed as v_t[B][N][H][L] for the attn.v GEMM.
// ---------------------------------------------------------------------------
__global__ __launch_bounds__(256) void k_proj(
    const __bf16* __restrict__ X, const __bf16* __restrict__ Wt4,
    __bf16* __restrict__ u_bf, __bf16* __restrict__ q_bf,
    __bf16* __restrict__ k_bf, __bf16* __restrict__ v_t) {
  const int tid = threadIdx.x, lane = tid & 31, w = tid >> 5;
  const int wm = w & 3, wn = w >> 2;
  const int m0 = blockIdx.x * 128 + wm * 32;
  const int wsel = blockIdx.y >> 3;
  const int n0 = (blockIdx.y & 7) * 128 + wn * 64;
  const __bf16* W = Wt4 + (size_t)wsel * D_ * D_;

  v8f acc[2][4];
#pragma unroll
  for (int i = 0; i < 2; ++i)
#pragma unroll
    for (int j = 0; j < 4; ++j) acc[i][j] = vzero8();

  for (int k0 = 0; k0 < D_; k0 += 32) {
    v16bf a0 = load_frag_a(X + (size_t)m0 * D_ + k0, D_, lane);
    v16bf a1 = load_frag_a(X + (size_t)(m0 + 16) * D_ + k0, D_, lane);
#pragma unroll
    for (int j = 0; j < 4; ++j) {
      v16bf b = load_frag_b(W + (size_t)(n0 + j * 16) * D_ + k0, D_, lane);
      acc[0][j] = wmma_bf16(a0, b, acc[0][j]);
      acc[1][j] = wmma_bf16(a1, b, acc[1][j]);
    }
  }

  const int hi = lane >> 4, c15 = lane & 15;
#pragma unroll
  for (int i = 0; i < 2; ++i)
#pragma unroll
    for (int j = 0; j < 4; ++j) {
      v8f cc = acc[i][j];
#pragma unroll
      for (int e = 0; e < 8; ++e) {
        const int m = m0 + i * 16 + hi * 8 + e;
        const int n = n0 + j * 16 + c15;
        const __bf16 bv = (__bf16)silu_f(cc[e]);
        if (wsel == 0)      u_bf[(size_t)m * D_ + n] = bv;
        else if (wsel == 1) q_bf[(size_t)m * D_ + n] = bv;
        else if (wsel == 2) k_bf[(size_t)m * D_ + n] = bv;
        else {
          const int bb = m >> 11, t = m & (L_ - 1);
          const int head = n >> 7, h = n & (HD_ - 1);
          v_t[(((size_t)(bb * NH_ + head)) * HD_ + h) * L_ + t] = bv;
        }
      }
    }
}

// ---------------------------------------------------------------------------
// K3: causal pointwise-SiLU attention (no softmax -> plain GEMM-GEMM chain).
// Block = 128 t-rows of one (b, head); wave = 16-row t-band x full H=128.
// ---------------------------------------------------------------------------
__global__ __launch_bounds__(256) void k_attn(
    const __bf16* __restrict__ q_bf, const __bf16* __restrict__ k_bf,
    const __bf16* __restrict__ v_t, float* __restrict__ attn_out) {
  __shared__ __bf16 plds[8][16 * 32];
  const int tid = threadIdx.x, lane = tid & 31, w = tid >> 5;
  const int b  = blockIdx.y >> 3;
  const int nh = blockIdx.y & 7;
  const int t0 = blockIdx.x * 128 + w * 16;
  const int hi = lane >> 4, c15 = lane & 15;

  // q A-fragments for this t-band are loop-invariant over s.
  v16bf aQ[4];
#pragma unroll
  for (int kk = 0; kk < 4; ++kk)
    aQ[kk] = load_frag_a(q_bf + ((size_t)(b * L_ + t0)) * D_ + nh * HD_ + kk * 32,
                         D_, lane);

  v8f acc[8];
#pragma unroll
  for (int nt = 0; nt < 8; ++nt) acc[nt] = vzero8();

  const float inv_len = 1.0f / (float)L_;

  for (int s0 = 0; s0 <= t0 + 15; s0 += 32) {
    // ---- scores: 16 x 32 tile of silu(q.k)/L, causal-masked --------------
#pragma unroll
    for (int st = 0; st < 2; ++st) {
      v8f sc = vzero8();
      const __bf16* kb =
          k_bf + ((size_t)(b * L_ + s0 + st * 16)) * D_ + nh * HD_;
#pragma unroll
      for (int kk = 0; kk < 4; ++kk)
        sc = wmma_bf16(aQ[kk], load_frag_b(kb + kk * 32, D_, lane), sc);
#pragma unroll
      for (int e = 0; e < 8; ++e) {
        const int t = t0 + hi * 8 + e;
        const int s = s0 + st * 16 + c15;
        const float val = (s <= t) ? silu_f(sc[e]) * inv_len : 0.0f;
        plds[w][(hi * 8 + e) * 32 + st * 16 + c15] = (__bf16)val;
      }
    }
    lds_sync_wave();                       // C-layout -> A-layout via LDS
    v16bf aP = load_frag_a(&plds[w][0], 32, lane);
    lds_sync_wave();
    // ---- accumulate into output: P(16x32) x V(32x128) --------------------
#pragma unroll
    for (int nt = 0; nt < 8; ++nt) {
      const __bf16* vb =
          v_t + (((size_t)(b * NH_ + nh)) * HD_ + nt * 16) * L_ + s0;
      acc[nt] = wmma_bf16(aP, load_frag_b(vb, L_, lane), acc[nt]);
    }
  }

#pragma unroll
  for (int nt = 0; nt < 8; ++nt) {
    v8f cc = acc[nt];
#pragma unroll
    for (int e = 0; e < 8; ++e) {
      const int t = t0 + hi * 8 + e;
      attn_out[(((size_t)(b * L_ + t)) * NH_ + nh) * HD_ + nt * 16 + c15] = cc[e];
    }
  }
}

// ---------------------------------------------------------------------------
// K4: LayerNorm over (N,H)=1024 of attn_out, times u  -> bf16 g.
// ---------------------------------------------------------------------------
__global__ __launch_bounds__(256) void k_lnu(
    const float* __restrict__ ao, const float* __restrict__ gamma,
    const float* __restrict__ beta, const __bf16* __restrict__ u_bf,
    __bf16* __restrict__ g_bf) {
  const int m = blockIdx.x;
  const int tid = threadIdx.x, lane = tid & 31, w = tid >> 5;
  const float4 v = reinterpret_cast<const float4*>(ao + (size_t)m * D_)[tid];
  float s  = v.x + v.y + v.z + v.w;
  float sq = v.x * v.x + v.y * v.y + v.z * v.z + v.w * v.w;
  s = wave_sum(s); sq = wave_sum(sq);
  __shared__ float rs[8], rq[8];
  if (lane == 0) { rs[w] = s; rq[w] = sq; }
  __syncthreads();
  float tot = 0.f, totq = 0.f;
#pragma unroll
  for (int i = 0; i < 8; ++i) { tot += rs[i]; totq += rq[i]; }
  const float mean = tot * (1.0f / D_);
  const float rstd = rsqrtf(totq * (1.0f / D_) - mean * mean + EPS_);
  const float vals[4] = {v.x, v.y, v.z, v.w};
#pragma unroll
  for (int e = 0; e < 4; ++e) {
    const int i = tid * 4 + e;
    const float nrm = (vals[e] - mean) * rstd * gamma[i] + beta[i];
    g_bf[(size_t)m * D_ + i] = (__bf16)((float)u_bf[(size_t)m * D_ + i] * nrm);
  }
}

// ---------------------------------------------------------------------------
// K5: out = g @ wo + inputs   (f32 output)
// ---------------------------------------------------------------------------
__global__ __launch_bounds__(256) void k_outproj(
    const __bf16* __restrict__ G, const __bf16* __restrict__ WoT,
    const float* __restrict__ resid, float* __restrict__ out) {
  const int tid = threadIdx.x, lane = tid & 31, w = tid >> 5;
  const int wm = w & 3, wn = w >> 2;
  const int m0 = blockIdx.x * 128 + wm * 32;
  const int n0 = blockIdx.y * 128 + wn * 64;

  v8f acc[2][4];
#pragma unroll
  for (int i = 0; i < 2; ++i)
#pragma unroll
    for (int j = 0; j < 4; ++j) acc[i][j] = vzero8();

  for (int k0 = 0; k0 < D_; k0 += 32) {
    v16bf a0 = load_frag_a(G + (size_t)m0 * D_ + k0, D_, lane);
    v16bf a1 = load_frag_a(G + (size_t)(m0 + 16) * D_ + k0, D_, lane);
#pragma unroll
    for (int j = 0; j < 4; ++j) {
      v16bf b = load_frag_b(WoT + (size_t)(n0 + j * 16) * D_ + k0, D_, lane);
      acc[0][j] = wmma_bf16(a0, b, acc[0][j]);
      acc[1][j] = wmma_bf16(a1, b, acc[1][j]);
    }
  }

  const int hi = lane >> 4, c15 = lane & 15;
#pragma unroll
  for (int i = 0; i < 2; ++i)
#pragma unroll
    for (int j = 0; j < 4; ++j) {
      v8f cc = acc[i][j];
#pragma unroll
      for (int e = 0; e < 8; ++e) {
        const int m = m0 + i * 16 + hi * 8 + e;
        const int n = n0 + j * 16 + c15;
        out[(size_t)m * D_ + n] = cc[e] + resid[(size_t)m * D_ + n];
      }
    }
}

// ---------------------------------------------------------------------------
extern "C" void kernel_launch(void* const* d_in, const int* in_sizes, int n_in,
                              void* d_out, int out_size, void* d_ws, size_t ws_size,
                              hipStream_t stream) {
  (void)in_sizes; (void)n_in; (void)out_size; (void)ws_size;
  const float* inputs   = (const float*)d_in[0];
  // d_in[1] = attention_mask (causal tril) -- implemented analytically
  const float* ln_in_g  = (const float*)d_in[2];
  const float* ln_in_b  = (const float*)d_in[3];
  const float* wu       = (const float*)d_in[4];
  const float* wq       = (const float*)d_in[5];
  const float* wk       = (const float*)d_in[6];
  const float* wv       = (const float*)d_in[7];
  const float* ln_at_g  = (const float*)d_in[8];
  const float* ln_at_b  = (const float*)d_in[9];
  const float* wo       = (const float*)d_in[10];
  float* out = (float*)d_out;

  char* ws = (char*)d_ws;
  size_t off = 0;
  const size_t SZ_BF_MD = (size_t)M_ * D_ * sizeof(__bf16);   // 16 MiB
  __bf16* x_bf  = (__bf16*)(ws + off); off += SZ_BF_MD;
  __bf16* wt4   = (__bf16*)(ws + off); off += (size_t)4 * D_ * D_ * sizeof(__bf16);
  __bf16* wo_t  = (__bf16*)(ws + off); off += (size_t)D_ * D_ * sizeof(__bf16);
  __bf16* u_bf  = (__bf16*)(ws + off); off += SZ_BF_MD;
  __bf16* q_bf  = (__bf16*)(ws + off); off += SZ_BF_MD;
  __bf16* k_bf  = (__bf16*)(ws + off); off += SZ_BF_MD;
  __bf16* v_t   = (__bf16*)(ws + off); off += SZ_BF_MD;
  float*  ao    = (float*) (ws + off); off += (size_t)M_ * D_ * sizeof(float);
  __bf16* g_bf  = (__bf16*)(ws + off); off += SZ_BF_MD;

  // K0: weight transpose/convert  (5 matrices x 1M elements)
  k_wconv<<<dim3(4096, 5), 256, 0, stream>>>(wu, wq, wk, wv, wo, wt4, wo_t);
  // K1: input LN -> bf16
  k_ln_in<<<dim3(M_), 256, 0, stream>>>(inputs, ln_in_g, ln_in_b, x_bf);
  // K2: u/q/k/v projections (WMMA)
  k_proj<<<dim3(M_ / 128, 32), 256, 0, stream>>>(x_bf, wt4, u_bf, q_bf, k_bf, v_t);
  // K3: causal silu attention (WMMA x2 chain)
  k_attn<<<dim3(L_ / 128, B_ * NH_), 256, 0, stream>>>(q_bf, k_bf, v_t, ao);
  // K4: attn LN * u -> g
  k_lnu<<<dim3(M_), 256, 0, stream>>>(ao, ln_at_g, ln_at_b, u_bf, g_bf);
  // K5: output projection + residual (WMMA)
  k_outproj<<<dim3(M_ / 128, D_ / 128), 256, 0, stream>>>(g_bf, wo_t, inputs, out);
}
